// GCMC_40870908789353
// MI455X (gfx1250) — matrix operand order repaired and checked by reading, not compiled, not run on hardware
//
#include <hip/hip_runtime.h>
#include <hip/hip_bf16.h>
#include <hip/hip_fp16.h>

typedef _Float16 half_t;
typedef __attribute__((ext_vector_type(16))) _Float16 v16h;
typedef __attribute__((ext_vector_type(8)))  float    v8f;

#define NEG_SLOPE 0.01f

__device__ __forceinline__ float leakyf(float x) { return x > 0.f ? x : x * NEG_SLOPE; }

// ---------------- WMMA fragment helpers (CDNA5 16x16x32 f16, wave32) -------
// A-matrix 16x32 (MxK), ISA 7.12.2: lane holds row m = lane&15,
// K = kb+0..7 and kb+16..23 with kb = 8*(lane>>4) -> two contiguous 16B runs.
__device__ __forceinline__ v16h load_a_frag(const half_t* __restrict__ A, int lda,
                                            int m0, int k0) {
  const int lane = threadIdx.x & 31;
  const half_t* p = A + (size_t)(m0 + (lane & 15)) * lda + k0 + ((lane >> 4) << 3);
  v16h r;
  __builtin_memcpy(&r, p, 16);                    // K = kb+0..7   (global_load_b128)
  __builtin_memcpy((char*)&r + 16, p + 16, 16);   // K = kb+16..23 (global_load_b128)
  return r;
}

// B pre-swizzled into fragment order: tile t = nt*KT+kt, per lane 16 contiguous
// halfs; element j <-> B[32*kt + 8*(lane>>4) + (j&7) + 16*(j>>3)][16*nt + (lane&15)].
__device__ __forceinline__ v16h load_b_sw(const half_t* __restrict__ Wsw, int tile) {
  const int lane = threadIdx.x & 31;
  const half_t* p = Wsw + ((size_t)tile * 32 + lane) * 16;
  v16h r;
  __builtin_memcpy(&r, p, 32);                    // 2x global_load_b128, contiguous
  return r;
}

__device__ __forceinline__ v8f wmma16(v16h a, v16h b, v8f c) {
  return __builtin_amdgcn_wmma_f32_16x16x32_f16(false, a, false, b, (short)0, c,
                                                false, false);
}

// ---------------- elementwise / staging kernels -----------------------------

__global__ void k_normalize(const float* __restrict__ emb, half_t* __restrict__ out,
                            int n) {
  int r = blockIdx.x * blockDim.x + threadIdx.x;
  if (r >= n) return;
  const float4* p = (const float4*)(emb + (size_t)r * 64);
  float ss = 0.f;
#pragma unroll
  for (int i = 0; i < 16; ++i) {
    float4 v = p[i];
    ss += v.x * v.x + v.y * v.y + v.z * v.z + v.w * v.w;
  }
  float inv = 1.f / fmaxf(sqrtf(ss), 1e-12f);
  half_t* o = out + (size_t)r * 64;
#pragma unroll
  for (int i = 0; i < 16; ++i) {
    float4 v = p[i];
    o[4 * i + 0] = (half_t)(v.x * inv);
    o[4 * i + 1] = (half_t)(v.y * inv);
    o[4 * i + 2] = (half_t)(v.z * inv);
    o[4 * i + 3] = (half_t)(v.w * inv);
  }
}

__global__ void k_cvt_f32_f16(const float* __restrict__ s, half_t* __restrict__ d,
                              size_t n) {
  size_t i = (size_t)blockIdx.x * blockDim.x + threadIdx.x;
  if (i < n) d[i] = (half_t)s[i];
}

// Convert weight [K,64] f32 -> f16 in WMMA B-fragment order (see load_b_sw).
__global__ void k_cvt_wB(const float* __restrict__ W, half_t* __restrict__ out, int K) {
  int o = blockIdx.x * blockDim.x + threadIdx.x;
  if (o >= K * 64) return;
  const int KT = K >> 5;
  const int j = o & 15;
  const int lane = (o >> 4) & 31;
  const int t = o >> 9;
  const int kt = t % KT;
  const int nt = t / KT;
  const int k = 32 * kt + ((lane >> 4) << 3) + (j & 7) + ((j >> 3) << 4);
  const int n = 16 * nt + (lane & 15);
  out[o] = (half_t)W[k * 64 + n];
}

__global__ void k_leaky_to_h(const float* __restrict__ in, half_t* __restrict__ out,
                             size_t n) {
  size_t i = (size_t)blockIdx.x * blockDim.x + threadIdx.x;
  if (i < n) out[i] = (half_t)leakyf(in[i]);
}

// ---------------- GEMM kernels: 1 wave = 16x64 output stripe ---------------

// Out[rows,64] = A[rows,64](f16) @ W(swizzled, K=64)   (fp32 out)
__global__ void k_gemm64(const half_t* __restrict__ A, const half_t* __restrict__ Wsw,
                         float* __restrict__ Out, int rows) {
  const int wid = blockIdx.x * (blockDim.x >> 5) + (threadIdx.x >> 5);
  const int m0 = wid * 16;
  if (m0 >= rows) return;
  v8f acc[4] = {};
#pragma unroll
  for (int kt = 0; kt < 2; ++kt) {
    const v16h a = load_a_frag(A, 64, m0, 32 * kt);
#pragma unroll
    for (int nt = 0; nt < 4; ++nt)
      acc[nt] = wmma16(a, load_b_sw(Wsw, nt * 2 + kt), acc[nt]);
  }
  const int lane = threadIdx.x & 31;
  const int rbase = m0 + ((lane >> 4) << 3);
#pragma unroll
  for (int nt = 0; nt < 4; ++nt) {
    const int col = nt * 16 + (lane & 15);
#pragma unroll
    for (int v = 0; v < 8; ++v)
      Out[(size_t)(rbase + v) * 64 + col] = acc[nt][v];
  }
}

// f = leaky(cat(v_feat,t_feat)[rows,256] @ lin_w(swizzled, K=256) + b) -> f16
__global__ void k_gemm_f(const half_t* __restrict__ vfeat, const half_t* __restrict__ tfeat,
                         const half_t* __restrict__ Wsw, const float* __restrict__ bias,
                         half_t* __restrict__ Out, int rows) {
  const int wid = blockIdx.x * (blockDim.x >> 5) + (threadIdx.x >> 5);
  const int m0 = wid * 16;
  if (m0 >= rows) return;
  v8f acc[4] = {};
#pragma unroll
  for (int kt = 0; kt < 8; ++kt) {
    const half_t* src = (kt < 4) ? vfeat : tfeat;
    const int ka = (kt < 4) ? (32 * kt) : (32 * kt - 128);
    const v16h a = load_a_frag(src, 128, m0, ka);
#pragma unroll
    for (int nt = 0; nt < 4; ++nt)
      acc[nt] = wmma16(a, load_b_sw(Wsw, nt * 8 + kt), acc[nt]);
  }
  const int lane = threadIdx.x & 31;
  const int rbase = m0 + ((lane >> 4) << 3);
#pragma unroll
  for (int nt = 0; nt < 4; ++nt) {
    const int col = nt * 16 + (lane & 15);
    const float b = bias[col];
#pragma unroll
    for (int v = 0; v < 8; ++v)
      Out[(size_t)(rbase + v) * 64 + col] = (half_t)leakyf(acc[nt][v] + b);
  }
}

// xfinal = leaky(x2[rows,64] @ weight_W(swizzled) + pad(f_hat))   (fp32 out)
__global__ void k_gemm_final(const half_t* __restrict__ A, const half_t* __restrict__ Wsw,
                             const float* __restrict__ fhat, float* __restrict__ Out,
                             int rows, int num_user) {
  const int wid = blockIdx.x * (blockDim.x >> 5) + (threadIdx.x >> 5);
  const int m0 = wid * 16;
  if (m0 >= rows) return;
  v8f acc[4] = {};
#pragma unroll
  for (int kt = 0; kt < 2; ++kt) {
    const v16h a = load_a_frag(A, 64, m0, 32 * kt);
#pragma unroll
    for (int nt = 0; nt < 4; ++nt)
      acc[nt] = wmma16(a, load_b_sw(Wsw, nt * 2 + kt), acc[nt]);
  }
  const int lane = threadIdx.x & 31;
  const int rbase = m0 + ((lane >> 4) << 3);
#pragma unroll
  for (int nt = 0; nt < 4; ++nt) {
    const int col = nt * 16 + (lane & 15);
#pragma unroll
    for (int v = 0; v < 8; ++v) {
      const int r = rbase + v;
      float t = acc[nt][v];
      if (r >= num_user) t += fhat[(size_t)(r - num_user) * 64 + col];
      Out[(size_t)r * 64 + col] = leakyf(t);
    }
  }
}

// ---------------- scatter kernels (HBM/atomic bound) ------------------------

__global__ void k_scatter_edges(const float* __restrict__ h, const int* __restrict__ src,
                                const int* __restrict__ dst, float* __restrict__ agg,
                                int ne) {
  long long tid = (long long)blockIdx.x * blockDim.x + threadIdx.x;
  if (tid >= (long long)ne * 16) return;
  const int e = (int)(tid >> 4);
  const int g = (int)(tid & 15);
  const int s = src[e], d = dst[e];
  const float4 v = *(const float4*)(h + (size_t)s * 64 + g * 4);
  float* o = agg + (size_t)d * 64 + g * 4;
  unsafeAtomicAdd(o + 0, v.x);
  unsafeAtomicAdd(o + 1, v.y);
  unsafeAtomicAdd(o + 2, v.z);
  unsafeAtomicAdd(o + 3, v.w);
}

__global__ void k_word_cnt(const int* __restrict__ items, float* __restrict__ cnt,
                           int nw) {
  int i = blockIdx.x * blockDim.x + threadIdx.x;
  if (i < nw) unsafeAtomicAdd(&cnt[items[i]], 1.0f);
}

__global__ void k_word_sum(const int* __restrict__ items, const int* __restrict__ wids,
                           const float* __restrict__ table, float* __restrict__ sums,
                           int nw) {
  long long tid = (long long)blockIdx.x * blockDim.x + threadIdx.x;
  if (tid >= (long long)nw * 32) return;
  const int w = (int)(tid >> 5);
  const int g = (int)(tid & 31);
  const int item = items[w], wid = wids[w];
  const float4 v = *(const float4*)(table + (size_t)wid * 128 + g * 4);
  float* o = sums + (size_t)item * 128 + g * 4;
  unsafeAtomicAdd(o + 0, v.x);
  unsafeAtomicAdd(o + 1, v.y);
  unsafeAtomicAdd(o + 2, v.z);
  unsafeAtomicAdd(o + 3, v.w);
}

__global__ void k_tfeat(const float* __restrict__ sums, const float* __restrict__ cnt,
                        half_t* __restrict__ out, int nitem) {
  size_t i = (size_t)blockIdx.x * blockDim.x + threadIdx.x;
  if (i >= (size_t)nitem * 128) return;
  const int item = (int)(i >> 7);
  out[i] = (half_t)(sums[i] / fmaxf(cnt[item], 1.0f));
}

__global__ void k_scores(const float* __restrict__ X, const int* __restrict__ un,
                         const int* __restrict__ itn, float* __restrict__ out, int b) {
  int t = blockIdx.x * blockDim.x + threadIdx.x;
  if (t >= b) return;
  const float4* pu = (const float4*)(X + (size_t)un[t] * 64);
  const float4* pi = (const float4*)(X + (size_t)itn[t] * 64);
  float s = 0.f;
#pragma unroll
  for (int i = 0; i < 16; ++i) {
    float4 a = pu[i], c = pi[i];
    s += a.x * c.x + a.y * c.y + a.z * c.z + a.w * c.w;
  }
  out[t] = s;
}

// ---------------- launcher --------------------------------------------------

extern "C" void kernel_launch(void* const* d_in, const int* in_sizes, int n_in,
                              void* d_out, int out_size, void* d_ws, size_t ws_size,
                              hipStream_t stream) {
  const float* id_emb   = (const float*)d_in[0];
  const float* conv_w   = (const float*)d_in[1];
  const float* word_tab = (const float*)d_in[2];
  const float* v_feat   = (const float*)d_in[3];
  const float* lin_w    = (const float*)d_in[4];
  const float* lin_b    = (const float*)d_in[5];
  const float* weight_W = (const float*)d_in[6];
  const float* weight_2 = (const float*)d_in[7];
  const int*   edges    = (const int*)d_in[8];
  const int*   words    = (const int*)d_in[9];
  const int*   unodes   = (const int*)d_in[10];
  const int*   inodes   = (const int*)d_in[11];
  float* out = (float*)d_out;

  const int N     = in_sizes[0] / 64;     // 70000
  const int NITEM = in_sizes[3] / 128;    // 20000
  const int NUSER = N - NITEM;            // 50000
  const int NE2   = in_sizes[8] / 2;      // 2,000,000
  const int NW    = in_sizes[9] / 2;      // 1,000,000
  const int NB    = in_sizes[10];         // 8192

  // ---- workspace layout (buffer reuse, ~58 MB) ----
  char* ws = (char*)d_ws;
  size_t off = 0;
  auto carve = [&](size_t bytes) { size_t o = off; off += (bytes + 255) & ~(size_t)255; return o; };
  half_t* xnorm_h = (half_t*)(ws + carve((size_t)N * 64 * 2));      // later reused as x2h
  float*  bufB    = (float*)(ws + carve((size_t)N * 64 * 4));       // agg, later xfinal
  float*  bufC    = (float*)(ws + carve((size_t)N * 64 * 4));       // h, later sums/cnt/tfeat
  half_t* vfeat_h = (half_t*)(ws + carve((size_t)NITEM * 128 * 2));
  half_t* f_h     = (half_t*)(ws + carve((size_t)NITEM * 64 * 2));
  float*  fhat    = (float*)(ws + carve((size_t)NITEM * 64 * 4));
  half_t* convW_s = (half_t*)(ws + carve(64 * 64 * 2));
  half_t* wW_s    = (half_t*)(ws + carve(64 * 64 * 2));
  half_t* w2_s    = (half_t*)(ws + carve(64 * 64 * 2));
  half_t* linw_s  = (half_t*)(ws + carve(256 * 64 * 2));

  half_t* x2h     = xnorm_h;                     // reuse after gemm1
  float*  agg     = bufB;
  float*  xfinal  = bufB;                        // reuse after leaky
  float*  h       = bufC;
  float*  sums    = bufC;                        // reuse after edge scatter
  float*  cnt     = (float*)((char*)bufC + (((size_t)NITEM * 128 * 4 + 255) & ~(size_t)255));
  half_t* tfeat_h = (half_t*)((char*)cnt + (((size_t)NITEM * 4 + 255) & ~(size_t)255));

  const int TB = 256;
  const int WPB = TB / 32;  // waves (16-row stripes) per block in GEMM kernels

  // 1. normalize + weight conversion (swizzled to WMMA fragment order)
  k_normalize<<<(N + TB - 1) / TB, TB, 0, stream>>>(id_emb, xnorm_h, N);
  k_cvt_wB<<<(4096 + TB - 1) / TB, TB, 0, stream>>>(conv_w, convW_s, 64);
  k_cvt_wB<<<(4096 + TB - 1) / TB, TB, 0, stream>>>(weight_W, wW_s, 64);
  k_cvt_wB<<<(4096 + TB - 1) / TB, TB, 0, stream>>>(weight_2, w2_s, 64);
  k_cvt_wB<<<(16384 + TB - 1) / TB, TB, 0, stream>>>(lin_w, linw_s, 256);
  k_cvt_f32_f16<<<((size_t)NITEM * 128 + TB - 1) / TB, TB, 0, stream>>>(v_feat, vfeat_h,
                                                                        (size_t)NITEM * 128);
  // 2. h = xnorm @ conv_weight  (WMMA)
  {
    int tiles = N / 16;
    k_gemm64<<<(tiles + WPB - 1) / WPB, TB, 0, stream>>>(xnorm_h, convW_s, h, N);
  }
  // 3. edge scatter-add (dominant cost: ~1 GB random traffic)
  hipMemsetAsync(agg, 0, (size_t)N * 64 * 4, stream);
  {
    long long tot = (long long)NE2 * 16;
    k_scatter_edges<<<(unsigned)((tot + TB - 1) / TB), TB, 0, stream>>>(h, edges, edges + NE2,
                                                                        agg, NE2);
  }
  // 4. x2 = leaky(agg) -> f16; frees h for reuse
  k_leaky_to_h<<<((size_t)N * 64 + TB - 1) / TB, TB, 0, stream>>>(agg, x2h, (size_t)N * 64);

  // 5. word scatter-mean
  hipMemsetAsync(sums, 0, (size_t)NITEM * 128 * 4, stream);
  hipMemsetAsync(cnt, 0, (size_t)NITEM * 4, stream);
  k_word_cnt<<<(NW + TB - 1) / TB, TB, 0, stream>>>(words, cnt, NW);
  {
    long long tot = (long long)NW * 32;
    k_word_sum<<<(unsigned)((tot + TB - 1) / TB), TB, 0, stream>>>(words, words + NW,
                                                                   word_tab, sums, NW);
  }
  k_tfeat<<<((size_t)NITEM * 128 + TB - 1) / TB, TB, 0, stream>>>(sums, cnt, tfeat_h, NITEM);

  // 6. f = leaky(cat(v,t) @ lin_w + b)  (WMMA, K=256)
  {
    int tiles = NITEM / 16;
    k_gemm_f<<<(tiles + WPB - 1) / WPB, TB, 0, stream>>>(vfeat_h, tfeat_h, linw_s, lin_b,
                                                         f_h, NITEM);
    // 7. f_hat = f @ weight_2  (WMMA)
    k_gemm64<<<(tiles + WPB - 1) / WPB, TB, 0, stream>>>(f_h, w2_s, fhat, NITEM);
  }
  // 8. xfinal = leaky(x2 @ weight_W + pad(f_hat))  (WMMA)
  {
    int tiles = N / 16;
    k_gemm_final<<<(tiles + WPB - 1) / WPB, TB, 0, stream>>>(x2h, wW_s, fhat, xfinal,
                                                             N, NUSER);
  }
  // 9. scores
  k_scores<<<(NB + TB - 1) / TB, TB, 0, stream>>>(xfinal, unodes, inodes, out, NB);
}
